// GatNet_75625784148569
// MI455X (gfx1250) — compile-verified, matrix operand order checked by reading, not run on hardware
//
#include <hip/hip_runtime.h>
#include <hip/hip_bf16.h>
#include <stdint.h>

// ---------------- problem constants ----------------
#define NN      100000        // nodes
#define ER      3200000       // raw edges
#define ET      (ER + NN)     // edges incl. self loops
#define FIN     256           // input features
#define HC1     64            // H1*C1
#define NH1     8             // heads layer1
#define NC1     8             // channels/head layer1
#define NCLS    41            // classes
#define NCLSP   48            // padded to 3x16 tiles
#define NEG_SLOPE 0.2f

typedef __attribute__((ext_vector_type(16))) _Float16 v16h;
typedef __attribute__((ext_vector_type(8)))  _Float16 v8h;
typedef __attribute__((ext_vector_type(8)))  float    v8f;

// monotonic uint encoding for float atomic-max
__device__ __forceinline__ unsigned fenc(float f) {
  unsigned u = __float_as_uint(f);
  return (u & 0x80000000u) ? ~u : (u | 0x80000000u);
}
__device__ __forceinline__ float fdec(unsigned u) {
  return (u & 0x80000000u) ? __uint_as_float(u & 0x7FFFFFFFu) : __uint_as_float(~u);
}
__device__ __forceinline__ float lrelu(float x) { return x > 0.f ? x : NEG_SLOPE * x; }

// edge e -> (src,dst), self-loops appended after raw edges
__device__ __forceinline__ void edge_sd(long long e, const long long* __restrict__ ei,
                                        int& s, int& d) {
  if (e < ER) { s = (int)ei[e]; d = (int)ei[ER + e]; }
  else        { s = d = (int)(e - ER); }
}

// ---------------- utility kernels ----------------
__global__ void k_zero(uint32_t* __restrict__ p, size_t n) {
  size_t i = (size_t)blockIdx.x * blockDim.x + threadIdx.x;
  size_t st = (size_t)gridDim.x * blockDim.x;
  for (; i < n; i += st) p[i] = 0u;
}

__global__ void k_cvt_x(const float* __restrict__ x, _Float16* __restrict__ xh) {
  size_t t = (size_t)blockIdx.x * blockDim.x + threadIdx.x;
  if (t < (size_t)NN * FIN) xh[t] = (_Float16)x[t];
}

__global__ void k_prep_w1t(const float* __restrict__ W1, _Float16* __restrict__ W1T) {
  int t = blockIdx.x * blockDim.x + threadIdx.x;
  if (t >= HC1 * FIN) return;
  int c = t >> 8, k = t & 255;               // W1T[c][k] = W1[k][c]
  W1T[t] = (_Float16)W1[k * HC1 + c];
}

__global__ void k_prep_w2t(const float* __restrict__ W2, _Float16* __restrict__ W2T) {
  int t = blockIdx.x * blockDim.x + threadIdx.x;
  if (t >= NCLSP * HC1) return;
  int c = t / HC1, k = t % HC1;              // W2T[c][k] = W2[k][c], pad cols 41..47
  W2T[t] = (_Float16)((c < NCLS) ? W2[k * NCLS + c] : 0.0f);
}

// ---------------- WMMA GEMM: C[M,N] = A[M,K] * BT[N,K]^T  (f16 in, f32 out) ----------------
// one wave per 16x16 tile; M multiple of 16, N multiple of 16, K multiple of 32
__global__ void k_gemm_wmma(const _Float16* __restrict__ A, const _Float16* __restrict__ BT,
                            float* __restrict__ C, int M, int K, int N) {
  const int lane = threadIdx.x & 31;
  const int wave = blockIdx.x * (blockDim.x >> 5) + (threadIdx.x >> 5);
  const int ntiles = N >> 4;
  const int mt = wave / ntiles;
  const int nt = wave - mt * ntiles;
  if (mt * 16 >= M) return;                  // wave-uniform guard: EXEC all-ones inside

  const int r  = lane & 15;
  const int kh = (lane >> 4) << 3;           // K sub-offset 0 or 8 per ISA 16-bit A layout
  const _Float16* ap = A  + (size_t)(mt * 16 + r) * K + kh;
  const _Float16* bp = BT + (size_t)(nt * 16 + r) * K + kh;

  v8f acc = {};
  for (int k0 = 0; k0 < K; k0 += 32) {
    v8h a0 = *(const v8h*)(ap + k0);         // K = kh+0..7
    v8h a1 = *(const v8h*)(ap + k0 + 16);    // K = kh+16..23
    v8h b0 = *(const v8h*)(bp + k0);
    v8h b1 = *(const v8h*)(bp + k0 + 16);
    v16h a, b;
#pragma unroll
    for (int i = 0; i < 8; ++i) { a[i] = a0[i]; a[8 + i] = a1[i];
                                  b[i] = b0[i]; b[8 + i] = b1[i]; }
    acc = __builtin_amdgcn_wmma_f32_16x16x32_f16(false, a, false, b,
                                                 (short)0, acc, false, false);
  }
  // C/D layout: VGPR j -> row j (lanes 0-15) / row j+8 (lanes 16-31), col = lane&15
  const int crow = mt * 16 + ((lane >> 4) << 3);
  float* cp = C + (size_t)crow * N + nt * 16 + (lane & 15);
#pragma unroll
  for (int j = 0; j < 8; ++j) cp[(size_t)j * N] = acc[j];
}

// ---------------- attention coefficients ----------------
__global__ void k_attn1(const float* __restrict__ h1, const float* __restrict__ a_src,
                        const float* __restrict__ a_dst,
                        float* __restrict__ s, float* __restrict__ d) {
  long long t = (long long)blockIdx.x * blockDim.x + threadIdx.x;
  if (t >= (long long)NN * NH1) return;
  int n = (int)(t >> 3), h = (int)(t & 7);
  const float* hp = h1 + (size_t)n * HC1 + h * NC1;
  float ss = 0.f, dd = 0.f;
#pragma unroll
  for (int c = 0; c < NC1; ++c) { ss += hp[c] * a_src[h * NC1 + c];
                                  dd += hp[c] * a_dst[h * NC1 + c]; }
  s[t] = ss; d[t] = dd;
}

__global__ void k_attn2(const float* __restrict__ h3, const float* __restrict__ a_src,
                        const float* __restrict__ a_dst,
                        float* __restrict__ s, float* __restrict__ d) {
  int n = blockIdx.x * blockDim.x + threadIdx.x;
  if (n >= NN) return;
  const float* hp = h3 + (size_t)n * NCLSP;
  float ss = 0.f, dd = 0.f;
  for (int c = 0; c < NCLS; ++c) { float v = hp[c]; ss += v * a_src[c]; dd += v * a_dst[c]; }
  s[n] = ss; d[n] = dd;
}

// ---------------- edge passes (layer 1: per edge*head) ----------------
__global__ void k_edge_max1(const long long* __restrict__ ei, const float* __restrict__ s1,
                            const float* __restrict__ d1, unsigned* __restrict__ m1) {
  long long t = (long long)blockIdx.x * blockDim.x + threadIdx.x;
  if (t >= (long long)ET * NH1) return;
  long long e = t >> 3; int h = (int)(t & 7);
  int s, d; edge_sd(e, ei, s, d);
  float ev = lrelu(s1[(size_t)s * NH1 + h] + d1[(size_t)d * NH1 + h]);
  atomicMax(&m1[(size_t)d * NH1 + h], fenc(ev));
}

__global__ void k_edge_acc1(const long long* __restrict__ ei, const float* __restrict__ s1,
                            const float* __restrict__ d1, const unsigned* __restrict__ m1,
                            const float* __restrict__ h1,
                            float* __restrict__ z1, float* __restrict__ acc1) {
  long long t = (long long)blockIdx.x * blockDim.x + threadIdx.x;
  if (t >= (long long)ET * NH1) return;
  long long e = t >> 3; int h = (int)(t & 7);
  int s, d; edge_sd(e, ei, s, d);
  float ev = lrelu(s1[(size_t)s * NH1 + h] + d1[(size_t)d * NH1 + h]);
  float w = __expf(ev - fdec(m1[(size_t)d * NH1 + h]));
  atomicAdd(&z1[(size_t)d * NH1 + h], w);
  const float* hp = h1 + (size_t)s * HC1 + h * NC1;
  float* ap = acc1 + (size_t)d * HC1 + h * NC1;
#pragma unroll
  for (int c = 0; c < NC1; ++c) atomicAdd(&ap[c], w * hp[c]);
}

// normalize + bias + ELU -> f16 features for layer 2
__global__ void k_fin1(const float* __restrict__ acc1, const float* __restrict__ z1,
                       const float* __restrict__ b1, _Float16* __restrict__ h2h) {
  long long t = (long long)blockIdx.x * blockDim.x + threadIdx.x;
  if (t >= (long long)NN * HC1) return;
  int n = (int)(t >> 6), j = (int)(t & 63), h = j >> 3;
  float v = acc1[t] / (z1[(size_t)n * NH1 + h] + 1e-16f) + b1[j];
  v = (v > 0.f) ? v : (__expf(v) - 1.f);     // ELU
  h2h[t] = (_Float16)v;
}

// ---------------- edge passes (layer 2: per edge, 41 channels) ----------------
__global__ void k_edge_max2(const long long* __restrict__ ei, const float* __restrict__ s2,
                            const float* __restrict__ d2, unsigned* __restrict__ m2) {
  long long e = (long long)blockIdx.x * blockDim.x + threadIdx.x;
  if (e >= ET) return;
  int s, d; edge_sd(e, ei, s, d);
  atomicMax(&m2[d], fenc(lrelu(s2[s] + d2[d])));
}

__global__ void k_edge_acc2(const long long* __restrict__ ei, const float* __restrict__ s2,
                            const float* __restrict__ d2, const unsigned* __restrict__ m2,
                            const float* __restrict__ h3,
                            float* __restrict__ z2, float* __restrict__ acc2) {
  long long e = (long long)blockIdx.x * blockDim.x + threadIdx.x;
  if (e >= ET) return;
  int s, d; edge_sd(e, ei, s, d);
  float w = __expf(lrelu(s2[s] + d2[d]) - fdec(m2[d]));
  atomicAdd(&z2[d], w);
  const float* hp = h3 + (size_t)s * NCLSP;
  float* ap = acc2 + (size_t)d * NCLS;
  for (int c = 0; c < NCLS; ++c) atomicAdd(&ap[c], w * hp[c]);
}

// normalize + bias + log_softmax -> output
__global__ void k_fin2(const float* __restrict__ acc2, const float* __restrict__ z2,
                       const float* __restrict__ b2, float* __restrict__ out) {
  int n = blockIdx.x * blockDim.x + threadIdx.x;
  if (n >= NN) return;
  float zi = 1.f / (z2[n] + 1e-16f);
  const float* a = acc2 + (size_t)n * NCLS;
  float mx = -3.0e38f;
  for (int c = 0; c < NCLS; ++c) { float l = a[c] * zi + b2[c]; mx = fmaxf(mx, l); }
  float sum = 0.f;
  for (int c = 0; c < NCLS; ++c) { float l = a[c] * zi + b2[c]; sum += __expf(l - mx); }
  float lse = mx + __logf(sum);
  float* o = out + (size_t)n * NCLS;
  for (int c = 0; c < NCLS; ++c) o[c] = a[c] * zi + b2[c] - lse;
}

// ---------------- host launcher ----------------
extern "C" void kernel_launch(void* const* d_in, const int* in_sizes, int n_in,
                              void* d_out, int out_size, void* d_ws, size_t ws_size,
                              hipStream_t stream) {
  const float*     x     = (const float*)d_in[0];
  const long long* ei    = (const long long*)d_in[1];
  const float*     W1    = (const float*)d_in[2];
  const float*     a_s1  = (const float*)d_in[3];
  const float*     a_d1  = (const float*)d_in[4];
  const float*     b1    = (const float*)d_in[5];
  const float*     W2    = (const float*)d_in[6];
  const float*     a_s2  = (const float*)d_in[7];
  const float*     a_d2  = (const float*)d_in[8];
  const float*     b2    = (const float*)d_in[9];
  float* out = (float*)d_out;

  // carve workspace (256B aligned slabs)
  char* base = (char*)d_ws;
  size_t off = 0;
  auto carve = [&](size_t sz) { size_t o = off; off += (sz + 255) & ~(size_t)255; return o; };
  size_t o_xh   = carve((size_t)NN * FIN * 2);      // x in f16
  size_t o_w1t  = carve((size_t)HC1 * FIN * 2);     // W1^T f16
  size_t o_w2t  = carve((size_t)NCLSP * HC1 * 2);   // W2^T f16 (padded)
  size_t o_h1   = carve((size_t)NN * HC1 * 4);      // layer1 pre-attn features f32
  size_t o_s1   = carve((size_t)NN * NH1 * 4);
  size_t o_d1   = carve((size_t)NN * NH1 * 4);
  size_t o_h2h  = carve((size_t)NN * HC1 * 2);      // layer2 input f16
  size_t o_h3   = carve((size_t)NN * NCLSP * 4);    // layer2 pre-attn features f32
  size_t o_s2   = carve((size_t)NN * 4);
  size_t o_d2   = carve((size_t)NN * 4);
  // contiguous zero-init region
  size_t o_zb   = off;
  size_t o_m1   = carve((size_t)NN * NH1 * 4);
  size_t o_z1   = carve((size_t)NN * NH1 * 4);
  size_t o_a1   = carve((size_t)NN * HC1 * 4);
  size_t o_m2   = carve((size_t)NN * 4);
  size_t o_z2   = carve((size_t)NN * 4);
  size_t o_a2   = carve((size_t)NN * NCLS * 4);
  size_t zero_words = (off - o_zb) / 4;
  (void)ws_size; (void)n_in; (void)in_sizes; (void)out_size;

  _Float16* xh  = (_Float16*)(base + o_xh);
  _Float16* W1T = (_Float16*)(base + o_w1t);
  _Float16* W2T = (_Float16*)(base + o_w2t);
  float*    h1  = (float*)(base + o_h1);
  float*    s1  = (float*)(base + o_s1);
  float*    d1  = (float*)(base + o_d1);
  _Float16* h2h = (_Float16*)(base + o_h2h);
  float*    h3  = (float*)(base + o_h3);
  float*    s2  = (float*)(base + o_s2);
  float*    d2  = (float*)(base + o_d2);
  unsigned* m1  = (unsigned*)(base + o_m1);
  float*    z1  = (float*)(base + o_z1);
  float*    a1  = (float*)(base + o_a1);
  unsigned* m2  = (unsigned*)(base + o_m2);
  float*    z2  = (float*)(base + o_z2);
  float*    a2  = (float*)(base + o_a2);

  const int B = 256;
  // 0) zero accumulators (atomic-max encoding: 0 == minimum key, safe since every node self-loops)
  k_zero<<<4096, B, 0, stream>>>((uint32_t*)(base + o_zb), zero_words);
  // 1) precision convert / weight transpose
  k_cvt_x   <<<(int)(((size_t)NN * FIN + B - 1) / B), B, 0, stream>>>(x, xh);
  k_prep_w1t<<<(HC1 * FIN + B - 1) / B, B, 0, stream>>>(W1, W1T);
  k_prep_w2t<<<(NCLSP * HC1 + B - 1) / B, B, 0, stream>>>(W2, W2T);
  // 2) layer-1 GEMM on WMMA: h1[NN,64] = xh[NN,256] @ W1
  {
    int waves = (NN / 16) * (HC1 / 16);
    k_gemm_wmma<<<(waves + 7) / 8, B, 0, stream>>>(xh, W1T, h1, NN, FIN, HC1);
  }
  // 3) attention coefficients, edge softmax (max pass + fused exp/sum/aggregate pass)
  k_attn1<<<(int)(((size_t)NN * NH1 + B - 1) / B), B, 0, stream>>>(h1, a_s1, a_d1, s1, d1);
  {
    long long t = (long long)ET * NH1;
    int g = (int)((t + B - 1) / B);
    k_edge_max1<<<g, B, 0, stream>>>(ei, s1, d1, m1);
    k_edge_acc1<<<g, B, 0, stream>>>(ei, s1, d1, m1, h1, z1, a1);
  }
  // 4) normalize + bias + ELU -> f16
  k_fin1<<<(int)(((size_t)NN * HC1 + B - 1) / B), B, 0, stream>>>(a1, z1, b1, h2h);
  // 5) layer-2 GEMM on WMMA: h3[NN,48] = h2h[NN,64] @ W2 (padded)
  {
    int waves = (NN / 16) * (NCLSP / 16);
    k_gemm_wmma<<<(waves + 7) / 8, B, 0, stream>>>(h2h, W2T, h3, NN, HC1, NCLSP);
  }
  // 6) layer-2 attention + edge softmax
  k_attn2<<<(NN + B - 1) / B, B, 0, stream>>>(h3, a_s2, a_d2, s2, d2);
  {
    int g = (int)(((long long)ET + B - 1) / B);
    k_edge_max2<<<g, B, 0, stream>>>(ei, s2, d2, m2);
    k_edge_acc2<<<g, B, 0, stream>>>(ei, s2, d2, m2, h3, z2, a2);
  }
  // 7) normalize + bias + log_softmax
  k_fin2<<<(NN + B - 1) / B, B, 0, stream>>>(a2, z2, b2, out);
}
